// ReciverModel_75187697484083
// MI455X (gfx1250) — matrix operand order repaired and checked by reading, not compiled, not run on hardware
//
#include <hip/hip_runtime.h>
#include <cstdint>
#include <cstddef>

// ---------------------------------------------------------------------------
// ViT decoder block for MI455X (gfx1250, wave32, WMMA, TDM).
// All GEMMs: bf16 operands, f32 accumulate via v_wmma_f32_16x16x32_bf16.
// W tiles staged to LDS with tensor_load_to_lds (double-buffered, TENSORcnt).
// Roofline: ~30 GFLOP, ~65 MB HBM traffic (everything else L2-resident).
// ---------------------------------------------------------------------------

#define DMODEL  256
#define HEADS   8
#define DKH     32
#define SEQ     197
#define SP      224          // padded sequence: 14 tiles of 16
#define BATCH   64
#define NT      (BATCH * SP) // 14336 padded token rows (multiple of 32)
#define FFDIM   1024
#define PATCHES 196
#define KCONV   768          // 3*16*16

typedef __attribute__((ext_vector_type(16))) __bf16 v16bf;
typedef __attribute__((ext_vector_type(8)))  __bf16 v8bf;
typedef __attribute__((ext_vector_type(8)))  float  v8f;

static __device__ __forceinline__ v8f wmma_bf16(v16bf a, v16bf b, v8f c) {
  return __builtin_amdgcn_wmma_f32_16x16x32_bf16(false, a, false, b, (short)0, c,
                                                 false, false);
}
static __device__ __forceinline__ v16bf combine8(v8bf lo, v8bf hi) {
  return __builtin_shufflevector(lo, hi, 0, 1, 2, 3, 4, 5, 6, 7, 8, 9, 10, 11,
                                 12, 13, 14, 15);
}

// --------------------------- TDM support (gfx1250) -------------------------
#if defined(__has_builtin)
#if __has_builtin(__builtin_amdgcn_tensor_load_to_lds) && \
    __has_builtin(__builtin_amdgcn_s_wait_tensorcnt)
#define HAVE_TDM 1
#endif
#endif
#ifndef HAVE_TDM
#define HAVE_TDM 0
#endif

#if HAVE_TDM && defined(__HIP_DEVICE_COMPILE__)
typedef __attribute__((ext_vector_type(4))) unsigned int u32x4;
typedef __attribute__((ext_vector_type(8))) int          i32x8;
typedef __attribute__((ext_vector_type(4))) int          i32x4;

// 2-D tile load: tile_y rows x tile_x elements (bf16), row stride stride_x
// elements, into LDS byte offset lds_off. D# packing per CDNA5 ISA 8.3/8.4.
static __device__ __forceinline__ void tdm_load_2d(unsigned lds_off,
                                                   const void* gaddr,
                                                   unsigned tile_x,
                                                   unsigned tile_y,
                                                   unsigned tensor_x,
                                                   unsigned tensor_y,
                                                   unsigned long long stride_x) {
  const unsigned long long ga = (unsigned long long)(uintptr_t)gaddr;
  u32x4 g0;
  g0[0] = 1u;                                            // count=1 (valid D#)
  g0[1] = lds_off;                                       // lds_addr
  g0[2] = (unsigned)(ga & 0xffffffffu);                  // global_addr[31:0]
  g0[3] = (unsigned)((ga >> 32) & 0x01ffffffu) | (2u << 30);  // [56:32]|type=2
  i32x8 g1;
  g1[0] = (int)(1u << 16);                               // data_size=2B
  g1[1] = (int)((tensor_x & 0xffffu) << 16);             // tensor_dim0[15:0]
  g1[2] = (int)((tensor_x >> 16) | ((tensor_y & 0xffffu) << 16));
  g1[3] = (int)((tensor_y >> 16) | (tile_x << 16));      // tile_dim0
  g1[4] = (int)(tile_y & 0xffffu);                       // tile_dim1 (dim2=0)
  g1[5] = (int)(unsigned)(stride_x & 0xffffffffu);       // dim0_stride[31:0]
  g1[6] = (int)(unsigned)((stride_x >> 32) & 0xffffu);   // dim0_stride[47:32]
  g1[7] = 0;
  i32x4 z;
  z[0] = 0; z[1] = 0; z[2] = 0; z[3] = 0;
#if __clang_major__ >= 23
  i32x8 z8;
#pragma unroll
  for (int i = 0; i < 8; ++i) z8[i] = 0;
  __builtin_amdgcn_tensor_load_to_lds(g0, g1, z, z, z8, 0);
#else
  __builtin_amdgcn_tensor_load_to_lds(g0, g1, z, z, 0);
#endif
}
#endif  // HAVE_TDM device

// ---------------------------------------------------------------------------
// Generic GEMM: C[M,N] = A[M,K](bf16) * W[N,K](bf16)^T + bias, optional ReLU.
// One wave computes a 32x64 block of C.
// MODE: 0 = store f32, 1 = store bf16, 2 = store bf16 head-transposed
//       (Vt[b][h][dk][token], token-padded to SP) for the attention PV stage.
// W tiles (64 rows x 32 K) are staged to LDS by the Tensor Data Mover with
// double buffering when available; otherwise read directly (L2-resident).
// ---------------------------------------------------------------------------
template <int MODE, int RELU>
__global__ __launch_bounds__(32) void gemm_wmma(
    const __bf16* __restrict__ A, const __bf16* __restrict__ W,
    const float* __restrict__ bias, void* __restrict__ out, int N, int K) {
  extern __shared__ __bf16 smem[];  // 2 buffers x (64 x 32) bf16 = 8 KB
  const int lane = threadIdx.x;
  const int half = lane >> 4;
  const int r    = lane & 15;
  const int mbase = blockIdx.x * 32;
  const int nbase = blockIdx.y * 64;

  v8f acc[2][4];
#pragma unroll
  for (int i = 0; i < 2; ++i)
#pragma unroll
    for (int j = 0; j < 4; ++j)
#pragma unroll
      for (int e = 0; e < 8; ++e) acc[i][j][e] = 0.0f;

#if HAVE_TDM && defined(__HIP_DEVICE_COMPILE__)
  // ---- prologue: DMA first W tile ----
  tdm_load_2d(0, W + (size_t)nbase * K, 32, 64, (unsigned)K, 64,
              (unsigned long long)K);
  int buf = 0;
  for (int kt = 0; kt < K; kt += 32) {
    const __bf16* a0p = A + (size_t)(mbase + r) * K + kt + half * 8;
    const __bf16* a1p = a0p + (size_t)16 * K;
    if (kt + 32 < K) {
      tdm_load_2d((unsigned)((buf ^ 1) * 4096),
                  W + (size_t)nbase * K + kt + 32, 32, 64, (unsigned)K, 64,
                  (unsigned long long)K);
      __builtin_prefetch(a0p + 32, 0, 3);
      __builtin_amdgcn_s_wait_tensorcnt(1);
    } else {
      __builtin_amdgcn_s_wait_tensorcnt(0);
    }
    v16bf a0 = combine8(*(const v8bf*)a0p, *(const v8bf*)(a0p + 16));
    v16bf a1 = combine8(*(const v8bf*)a1p, *(const v8bf*)(a1p + 16));
    const __bf16* wt = smem + buf * 2048;
#pragma unroll
    for (int j = 0; j < 4; ++j) {
      const __bf16* wp = wt + (j * 16 + r) * 32 + half * 8;
      v16bf bf = combine8(*(const v8bf*)wp, *(const v8bf*)(wp + 16));
      acc[0][j] = wmma_bf16(a0, bf, acc[0][j]);
      acc[1][j] = wmma_bf16(a1, bf, acc[1][j]);
    }
    buf ^= 1;
  }
#else
  for (int kt = 0; kt < K; kt += 32) {
    const __bf16* a0p = A + (size_t)(mbase + r) * K + kt + half * 8;
    const __bf16* a1p = a0p + (size_t)16 * K;
    v16bf a0 = combine8(*(const v8bf*)a0p, *(const v8bf*)(a0p + 16));
    v16bf a1 = combine8(*(const v8bf*)a1p, *(const v8bf*)(a1p + 16));
#pragma unroll
    for (int j = 0; j < 4; ++j) {
      const __bf16* wp = W + (size_t)(nbase + j * 16 + r) * K + kt + half * 8;
      v16bf bf = combine8(*(const v8bf*)wp, *(const v8bf*)(wp + 16));
      acc[0][j] = wmma_bf16(a0, bf, acc[0][j]);
      acc[1][j] = wmma_bf16(a1, bf, acc[1][j]);
    }
  }
#endif

#pragma unroll
  for (int i = 0; i < 2; ++i)
#pragma unroll
    for (int j = 0; j < 4; ++j) {
      const int col = nbase + j * 16 + r;
      const float bv = bias ? bias[col] : 0.0f;
#pragma unroll
      for (int e = 0; e < 8; ++e) {
        const int row = mbase + i * 16 + 8 * half + e;  // C layout (ISA 7.12.2)
        float v = acc[i][j][e] + bv;
        if (RELU) v = fmaxf(v, 0.0f);
        if (MODE == 0) {
          ((float*)out)[(size_t)row * N + col] = v;
        } else if (MODE == 1) {
          ((__bf16*)out)[(size_t)row * N + col] = (__bf16)v;
        } else {  // head-transposed V: Vt[b][h][dk][token]
          const int b = row / SP, tok = row % SP;
          const int h = col / DKH, dd = col % DKH;
          ((__bf16*)out)[(((size_t)b * HEADS + h) * DKH + dd) * SP + tok] =
              (__bf16)v;
        }
      }
    }
}

// ---------------------------------------------------------------------------
// Attention core: one wave per (q-tile, head, batch). QK^T + softmax + PV.
// V is head-transposed (Vt[b][h][dk][token]) so the PV B-fragment is two
// 16-byte loads per lane (8 contiguous tokens each).
// ---------------------------------------------------------------------------
__global__ __launch_bounds__(32) void attn_core(
    const __bf16* __restrict__ Q, const __bf16* __restrict__ Kmat,
    const __bf16* __restrict__ Vt, __bf16* __restrict__ O,
    const int* __restrict__ mask, float scale) {
  __shared__ float  sc[16][SP];
  __shared__ __bf16 pr[16][SP];
  __shared__ float  rmax[16], rinv[16];

  const int lane = threadIdx.x;
  const int half = lane >> 4;
  const int r    = lane & 15;
  const int mt = blockIdx.x, h = blockIdx.y, b = blockIdx.z;
  const int mbase = mt * 16;
  const size_t base  = ((size_t)b * SP) * DMODEL + h * DKH;
  const size_t vbase = ((size_t)b * HEADS + h) * DKH * SP;

  // --- Q fragment (A-matrix 16x32, K = head dim = 32 exactly) ---
  const __bf16* qp = Q + base + (size_t)(mbase + r) * DMODEL + half * 8;
  v16bf qa = combine8(*(const v8bf*)qp, *(const v8bf*)(qp + 16));

  // --- scores: one WMMA per 16-key tile ---
  for (int nt = 0; nt < SP / 16; ++nt) {
    const __bf16* kp = Kmat + base + (size_t)(nt * 16 + r) * DMODEL + half * 8;
    v16bf kb = combine8(*(const v8bf*)kp, *(const v8bf*)(kp + 16));
    v8f c;
#pragma unroll
    for (int e = 0; e < 8; ++e) c[e] = 0.0f;
    c = wmma_bf16(qa, kb, c);
    const int col = nt * 16 + r;
#pragma unroll
    for (int e = 0; e < 8; ++e) sc[8 * half + e][col] = c[e] * scale;
  }
  __syncthreads();

  // --- per-row masking + max + sum (lane i owns query row i) ---
  if (lane < 16) {
    const int q = mbase + lane;
    float m = -3.0e38f;
    for (int k = 0; k < SP; ++k) {
      float v = sc[lane][k];
      bool ok = (k < SEQ) && (q >= SEQ || mask == nullptr || mask[q * SEQ + k] != 0);
      v = ok ? v : -1.0e30f;
      sc[lane][k] = v;
      m = fmaxf(m, v);
    }
    float s = 0.0f;
    for (int k = 0; k < SP; ++k) s += __expf(sc[lane][k] - m);
    rmax[lane] = m;
    rinv[lane] = 1.0f / s;
  }
  __syncthreads();

  // --- cooperative prob conversion to bf16 A-layout staging ---
  for (int t = lane; t < 16 * SP; t += 32) {
    const int rr = t / SP, k = t % SP;
    pr[rr][k] = (__bf16)(__expf(sc[rr][k] - rmax[rr]) * rinv[rr]);
  }
  __syncthreads();

  // --- PV: A = probs (16 x 32 keys), B = Vt (32 keys x 16 dims) x2 tiles ---
  v8f acc0, acc1;
#pragma unroll
  for (int e = 0; e < 8; ++e) { acc0[e] = 0.0f; acc1[e] = 0.0f; }
  for (int kc = 0; kc < SP / 32; ++kc) {
    const __bf16* pp = &pr[r][kc * 32 + half * 8];
    v16bf pa = combine8(*(const v8bf*)pp, *(const v8bf*)(pp + 16));
    const __bf16* vp = Vt + vbase + (size_t)r * SP + kc * 32 + half * 8;
    v16bf vb0 = combine8(*(const v8bf*)vp, *(const v8bf*)(vp + 16));
    v16bf vb1 = combine8(*(const v8bf*)(vp + (size_t)16 * SP),
                         *(const v8bf*)(vp + (size_t)16 * SP + 16));
    acc0 = wmma_bf16(pa, vb0, acc0);
    acc1 = wmma_bf16(pa, vb1, acc1);
  }
#pragma unroll
  for (int e = 0; e < 8; ++e) {
    const int row = mbase + 8 * half + e;
    O[base + (size_t)row * DMODEL + r]      = (__bf16)acc0[e];
    O[base + (size_t)row * DMODEL + 16 + r] = (__bf16)acc1[e];
  }
}

// ---------------------------------------------------------------------------
// Residual + LayerNorm: one wave (32 lanes x 8 values) per token row.
// ---------------------------------------------------------------------------
__global__ __launch_bounds__(32) void resid_ln_k(
    const float* __restrict__ y, const float* __restrict__ resid,
    int resid_padded, const float* __restrict__ g, const float* __restrict__ be,
    float* __restrict__ outf, __bf16* __restrict__ outb,
    float* __restrict__ dout) {
  const int row = blockIdx.x;
  const int b = row / SP, tok = row % SP;
  const int lane = threadIdx.x;
  const bool valid = tok < SEQ;

  float v[8];
  float sum = 0.0f;
  if (valid) {
    const float* yp = y + (size_t)row * DMODEL;
    const float* rp = resid + (size_t)(resid_padded ? row : (b * SEQ + tok)) * DMODEL;
#pragma unroll
    for (int i = 0; i < 8; ++i) {
      const int d = lane * 8 + i;
      v[i] = yp[d] + rp[d];
      sum += v[i];
    }
  } else {
#pragma unroll
    for (int i = 0; i < 8; ++i) v[i] = 0.0f;
  }
#pragma unroll
  for (int o = 16; o >= 1; o >>= 1) sum += __shfl_xor(sum, o, 32);
  const float mean = sum * (1.0f / DMODEL);
  float var = 0.0f;
#pragma unroll
  for (int i = 0; i < 8; ++i) { const float d0 = v[i] - mean; var += d0 * d0; }
#pragma unroll
  for (int o = 16; o >= 1; o >>= 1) var += __shfl_xor(var, o, 32);
  const float inv = rsqrtf(var * (1.0f / DMODEL) + 1e-5f);
#pragma unroll
  for (int i = 0; i < 8; ++i) {
    const int d = lane * 8 + i;
    const float o = valid ? ((v[i] - mean) * inv * g[d] + be[d]) : 0.0f;
    if (outf) outf[(size_t)row * DMODEL + d] = o;
    if (outb) outb[(size_t)row * DMODEL + d] = (__bf16)o;
    if (dout && valid) dout[((size_t)b * SEQ + tok) * DMODEL + d] = o;
  }
}

// ---------------------------------------------------------------------------
// Elementwise helpers.
// ---------------------------------------------------------------------------
__global__ __launch_bounds__(256) void f2bf_k(const float* __restrict__ s,
                                              __bf16* __restrict__ d, int n) {
  const int t = blockIdx.x * blockDim.x + threadIdx.x;
  if (t < n) d[t] = (__bf16)s[t];
}

__global__ __launch_bounds__(256) void im2col_k(const float* __restrict__ img,
                                                __bf16* __restrict__ out,
                                                int total) {
  const int t = blockIdx.x * blockDim.x + threadIdx.x;
  if (t >= total) return;
  const int col = t % KCONV;
  const int row = t / KCONV;
  const int p = row % PATCHES;
  const int b = row / PATCHES;
  const int c = col / 256, rem = col % 256, ph = rem / 16, pw = rem % 16;
  const int py = p / 14, px = p % 14;
  const float v =
      img[(((size_t)b * 3 + c) * 224 + (py * 16 + ph)) * 224 + (px * 16 + pw)];
  out[t] = (__bf16)v;
}

__global__ __launch_bounds__(256) void pack_dec_k(const float* __restrict__ x,
                                                  __bf16* __restrict__ out) {
  const int t = blockIdx.x * blockDim.x + threadIdx.x;
  if (t >= NT * DMODEL) return;
  const int d = t % DMODEL;
  const int row = t / DMODEL;
  const int b = row / SP, tok = row % SP;
  out[t] = (tok < SEQ) ? (__bf16)x[((size_t)b * SEQ + tok) * DMODEL + d]
                       : (__bf16)0.0f;
}

__global__ __launch_bounds__(256) void assemble_e_k(
    const float* __restrict__ tmp, const float* __restrict__ cls,
    const float* __restrict__ pos, __bf16* __restrict__ e) {
  const int t = blockIdx.x * blockDim.x + threadIdx.x;
  if (t >= NT * DMODEL) return;
  const int d = t % DMODEL;
  const int row = t / DMODEL;
  const int b = row / SP, tok = row % SP;
  float v;
  if (tok < PATCHES)
    v = tmp[((size_t)b * PATCHES + tok) * DMODEL + d] + pos[tok * DMODEL + d];
  else if (tok == PATCHES)  // cls appended at END (faithful to reference)
    v = cls[d] + pos[PATCHES * DMODEL + d];
  else
    v = 0.0f;
  e[t] = (__bf16)v;
}

// ---------------------------------------------------------------------------
// Launcher.
// ---------------------------------------------------------------------------
static inline size_t align256(size_t x) { return (x + 255) & ~(size_t)255; }

extern "C" void kernel_launch(void* const* d_in, const int* in_sizes, int n_in,
                              void* d_out, int out_size, void* d_ws,
                              size_t ws_size, hipStream_t stream) {
  (void)in_sizes; (void)n_in; (void)out_size; (void)ws_size;
  const float* enc_out = (const float*)d_in[0];
  const float* dec_in  = (const float*)d_in[1];
  const int*   mask    = (const int*)d_in[2];
  const float* conv_w  = (const float*)d_in[3];
  const float* conv_b  = (const float*)d_in[4];
  const float* cls     = (const float*)d_in[5];
  const float* pos     = (const float*)d_in[6];
  const float* ff_w1   = (const float*)d_in[7];
  const float* ff_b1   = (const float*)d_in[8];
  const float* ff_w2   = (const float*)d_in[9];
  const float* ff_b2   = (const float*)d_in[10];
  const float* ln1_g = (const float*)d_in[11], *ln1_b = (const float*)d_in[12];
  const float* ln2_g = (const float*)d_in[13], *ln2_b = (const float*)d_in[14];
  const float* ln3_g = (const float*)d_in[15], *ln3_b = (const float*)d_in[16];
  const float* a1_wq = (const float*)d_in[17], *a1_bq = (const float*)d_in[18];
  const float* a1_wk = (const float*)d_in[19], *a1_bk = (const float*)d_in[20];
  const float* a1_wv = (const float*)d_in[21], *a1_bv = (const float*)d_in[22];
  const float* a1_wo = (const float*)d_in[23], *a1_bo = (const float*)d_in[24];
  const float* a2_wq = (const float*)d_in[25], *a2_bq = (const float*)d_in[26];
  const float* a2_wk = (const float*)d_in[27], *a2_bk = (const float*)d_in[28];
  const float* a2_wv = (const float*)d_in[29], *a2_bv = (const float*)d_in[30];
  const float* a2_wo = (const float*)d_in[31], *a2_bo = (const float*)d_in[32];
  float* out = (float*)d_out;

  char* ws = (char*)d_ws;
  size_t off = 0;
  auto alloc = [&](size_t bytes) -> void* {
    void* p = ws + off;
    off = align256(off + bytes);
    return p;
  };
  const size_t WSZ = (size_t)DMODEL * DMODEL * sizeof(__bf16);
  __bf16* wq1b = (__bf16*)alloc(WSZ);
  __bf16* wk1b = (__bf16*)alloc(WSZ);
  __bf16* wv1b = (__bf16*)alloc(WSZ);
  __bf16* wo1b = (__bf16*)alloc(WSZ);
  __bf16* wq2b = (__bf16*)alloc(WSZ);
  __bf16* wk2b = (__bf16*)alloc(WSZ);
  __bf16* wv2b = (__bf16*)alloc(WSZ);
  __bf16* wo2b = (__bf16*)alloc(WSZ);
  __bf16* ffw1b  = (__bf16*)alloc((size_t)FFDIM * DMODEL * sizeof(__bf16));
  __bf16* ffw2b  = (__bf16*)alloc((size_t)DMODEL * FFDIM * sizeof(__bf16));
  __bf16* convwb = (__bf16*)alloc((size_t)DMODEL * KCONV * sizeof(__bf16));

  const size_t TOKB = (size_t)NT * DMODEL * sizeof(__bf16);
  const size_t TOKF = (size_t)NT * DMODEL * sizeof(float);
  __bf16* decinb = (__bf16*)alloc(TOKB);
  __bf16* eb     = (__bf16*)alloc(TOKB);
  __bf16* Qb     = (__bf16*)alloc(TOKB);
  __bf16* Kb     = (__bf16*)alloc(TOKB);
  __bf16* Vtb    = (__bf16*)alloc(TOKB);   // head-transposed V
  __bf16* Ob     = (__bf16*)alloc(TOKB);
  float*  x1f    = (float*)alloc(TOKF);
  __bf16* x1b    = (__bf16*)alloc(TOKB);
  float*  x2f    = (float*)alloc(TOKF);
  __bf16* x2b    = (__bf16*)alloc(TOKB);
  // arena0: im2col (19.3MB) then FFN hidden (29.4MB)
  void* arena0 = alloc((size_t)NT * FFDIM * sizeof(__bf16));
  // arena1: patch-GEMM f32 tmp (12.9MB) then projection/FFN f32 out (14.7MB)
  float* arena1 = (float*)alloc(TOKF);

  auto cvt = [&](const float* s, __bf16* d, int n) {
    f2bf_k<<<dim3((n + 255) / 256), dim3(256), 0, stream>>>(s, d, n);
  };
  cvt(a1_wq, wq1b, DMODEL * DMODEL);
  cvt(a1_wk, wk1b, DMODEL * DMODEL);
  cvt(a1_wv, wv1b, DMODEL * DMODEL);
  cvt(a1_wo, wo1b, DMODEL * DMODEL);
  cvt(a2_wq, wq2b, DMODEL * DMODEL);
  cvt(a2_wk, wk2b, DMODEL * DMODEL);
  cvt(a2_wv, wv2b, DMODEL * DMODEL);
  cvt(a2_wo, wo2b, DMODEL * DMODEL);
  cvt(ff_w1, ffw1b, FFDIM * DMODEL);
  cvt(ff_w2, ffw2b, DMODEL * FFDIM);
  cvt(conv_w, convwb, DMODEL * KCONV);

  const unsigned SMEM = 8192;  // double-buffered TDM W tiles

  // patch embed: im2col + GEMM [B*196,768] x [256,768]^T
  {
    const int total = BATCH * PATCHES * KCONV;
    im2col_k<<<dim3((total + 255) / 256), dim3(256), 0, stream>>>(
        enc_out, (__bf16*)arena0, total);
    gemm_wmma<0, 0><<<dim3(BATCH * PATCHES / 32, DMODEL / 64), dim3(32), SMEM,
                      stream>>>((const __bf16*)arena0, convwb, conv_b, arena1,
                                DMODEL, KCONV);
    const int tot2 = NT * DMODEL;
    assemble_e_k<<<dim3((tot2 + 255) / 256), dim3(256), 0, stream>>>(
        arena1, cls, pos, eb);
    pack_dec_k<<<dim3((tot2 + 255) / 256), dim3(256), 0, stream>>>(dec_in,
                                                                   decinb);
  }

  const dim3 gproj(NT / 32, DMODEL / 64), tb(32);
  const float scale = 0.17677669529663689f;  // 1/sqrt(32)

  // ---- self attention ----
  gemm_wmma<1, 0><<<gproj, tb, SMEM, stream>>>(decinb, wq1b, a1_bq, Qb, DMODEL, DMODEL);
  gemm_wmma<1, 0><<<gproj, tb, SMEM, stream>>>(decinb, wk1b, a1_bk, Kb, DMODEL, DMODEL);
  gemm_wmma<2, 0><<<gproj, tb, SMEM, stream>>>(decinb, wv1b, a1_bv, Vtb, DMODEL, DMODEL);
  attn_core<<<dim3(SP / 16, HEADS, BATCH), tb, 0, stream>>>(Qb, Kb, Vtb, Ob,
                                                            nullptr, scale);
  gemm_wmma<0, 0><<<gproj, tb, SMEM, stream>>>(Ob, wo1b, a1_bo, arena1, DMODEL, DMODEL);
  resid_ln_k<<<dim3(NT), tb, 0, stream>>>(arena1, dec_in, 0, ln1_g, ln1_b, x1f,
                                          x1b, nullptr);

  // ---- cross attention (q=k=e, v=dec_in, masked) ----
  gemm_wmma<1, 0><<<gproj, tb, SMEM, stream>>>(eb, wq2b, a2_bq, Qb, DMODEL, DMODEL);
  gemm_wmma<1, 0><<<gproj, tb, SMEM, stream>>>(eb, wk2b, a2_bk, Kb, DMODEL, DMODEL);
  gemm_wmma<2, 0><<<gproj, tb, SMEM, stream>>>(decinb, wv2b, a2_bv, Vtb, DMODEL, DMODEL);
  attn_core<<<dim3(SP / 16, HEADS, BATCH), tb, 0, stream>>>(Qb, Kb, Vtb, Ob,
                                                            mask, scale);
  gemm_wmma<0, 0><<<gproj, tb, SMEM, stream>>>(Ob, wo2b, a2_bo, arena1, DMODEL, DMODEL);
  resid_ln_k<<<dim3(NT), tb, 0, stream>>>(arena1, x1f, 1, ln2_g, ln2_b, x2f,
                                          x2b, nullptr);

  // ---- FFN ----
  gemm_wmma<1, 1><<<dim3(NT / 32, FFDIM / 64), tb, SMEM, stream>>>(
      x2b, ffw1b, ff_b1, arena0, FFDIM, DMODEL);
  gemm_wmma<0, 0><<<gproj, tb, SMEM, stream>>>((const __bf16*)arena0, ffw2b,
                                               ff_b2, arena1, DMODEL, FFDIM);
  resid_ln_k<<<dim3(NT), tb, 0, stream>>>(arena1, x2f, 1, ln3_g, ln3_b, nullptr,
                                          nullptr, out);
}
